// MorphSegModel_58016418234658
// MI455X (gfx1250) — matrix operand order repaired and checked by reading, not compiled, not run on hardware
//
#include <hip/hip_runtime.h>

// ---------------------------------------------------------------------------
// Problem constants (from the reference)
// ---------------------------------------------------------------------------
#define S_   2048
#define T_   1024
#define D_   768
#define H_   12
#define DH_  64
#define CV_  512
#define DC_  128
#define L_   16
#define CC_  8192
#define DF_  3072   // 4*D

// ---------------------------------------------------------------------------
// WMMA fragment types (CDNA5 wave32, v_wmma_f32_16x16x32_bf16)
// ---------------------------------------------------------------------------
typedef __attribute__((ext_vector_type(16))) __bf16 bf16x16;
typedef __attribute__((ext_vector_type(8)))  float  f32x8;

union FragAB { unsigned int u[8]; bf16x16 v; };
union FragC  { float f[8];        f32x8  v; };

__device__ __forceinline__ unsigned short f2bf(float f) {
    unsigned int u = __builtin_bit_cast(unsigned int, f);
    u += 0x7fffu + ((u >> 16) & 1u);          // round-to-nearest-even
    return (unsigned short)(u >> 16);
}

__device__ __forceinline__ float gelu_t(float x) {
    float x3 = x * x * x;
    return 0.5f * x * (1.f + tanhf(0.7978845608028654f * (x + 0.044715f * x3)));
}

// ---------------------------------------------------------------------------
// Generic tiled bf16 WMMA GEMM:  C = alpha * A(MxK) @ B(KxN) [+ Cin], act
// Template params:
//   TRANSB: B is transposed (element (k,n) at B[n*ldb + k])
//   NGUARD: N may be ragged (column mask); otherwise N % 128 == 0 assumed
// Runtime: optional f32 output Cout and/or bf16 output Cbf (same ldc),
//          act: 0 = none, 1 = gelu(tanh approx) applied before the writes.
//
// Requirements (all call sites satisfy them): K % 32 == 0, M % 128 == 0,
// lda/ldb even, bf16 buffers 4B-aligned.
//
// Block = 256 threads (8 waves). Tile 128x128, K-step 32, double-buffered LDS
// with a fetch -> compute -> commit software pipeline: global loads for tile
// kt+1 are issued BEFORE the WMMA block for tile kt and only committed to LDS
// after it, so s_wait_loadcnt lands behind the matrix math, not in front.
// LDS tiles are packed as (bf16,bf16) uint32 pairs so each fragment is
// two ds_load_b128 (A: K=half*8.., 16+half*8..; B: same as A).
// Templating keeps each instantiation's register footprint minimal (no spill).
// ---------------------------------------------------------------------------
template <bool TRANSB, bool NGUARD>
__global__ __launch_bounds__(256)
void k_gemm_bf16(const unsigned short* __restrict__ A, int lda,
                 const unsigned short* __restrict__ B, int ldb,
                 const float* __restrict__ Cin, float* __restrict__ Cout,
                 unsigned short* __restrict__ Cbf, int ldc,
                 int M, int N, int K, float alpha, int act) {
    __shared__ unsigned int Ap[2][128 * 16];
    __shared__ unsigned int Bp[2][128 * 16];

    const int tid  = threadIdx.x;
    const int row0 = blockIdx.y * 128;
    const int col0 = blockIdx.x * 128;
    const int wid  = tid >> 5, lane = tid & 31;
    const int wm   = wid >> 1, wn = wid & 1;
    const int half = lane >> 4, l16 = lane & 15;

    const unsigned int* A32 = (const unsigned int*)A;
    const unsigned int* B32 = (const unsigned int*)B;

    FragC acc[2][4];
#pragma unroll
    for (int mi = 0; mi < 2; ++mi)
#pragma unroll
        for (int ni = 0; ni < 4; ++ni)
#pragma unroll
            for (int r = 0; r < 8; ++r) acc[mi][ni].f[r] = 0.f;

    // ---- loop-invariant staging descriptors (computed once) ----
    // A: lds word index == p == tid + i*256 (r*16+kk == p); M % 128 == 0 -> no guard.
    int aoff[8];
#pragma unroll
    for (int i = 0; i < 8; ++i) {
        int p = tid + i * 256;
        aoff[i] = (((row0 + (p >> 4)) * lda) >> 1) + (p & 15);
    }
    int boff[8];
    unsigned int bmask[NGUARD ? 8 : 1];
#pragma unroll
    for (int i = 0; i < 8; ++i) {
        int p = tid + i * 256;
        int gn, kk;
        if (TRANSB) { kk = p & 15;  gn = col0 + (p >> 4); }
        else        { kk = p >> 7;  gn = col0 + (p & 127); }
        int cn = gn;
        if (NGUARD) {
            bool ok  = gn < N;
            cn       = ok ? gn : 0;
            bmask[i] = ok ? 0xFFFFFFFFu : 0u;
        }
        boff[i] = TRANSB ? (((cn * ldb) >> 1) + kk) : ((2 * kk) * ldb + cn);
    }

    unsigned int ra[8], rb0[8], rb1[TRANSB ? 1 : 8];

    auto fetch = [&](int k0) {             // issue global loads (no consumption)
        int ka = k0 >> 1;
#pragma unroll
        for (int i = 0; i < 8; ++i) ra[i] = A32[aoff[i] + ka];
        if (TRANSB) {
#pragma unroll
            for (int i = 0; i < 8; ++i) rb0[i] = B32[boff[i] + ka];
        } else {
            int kb = k0 * ldb;
#pragma unroll
            for (int i = 0; i < 8; ++i) {
                rb0[i] = B[boff[i] + kb];
                rb1[i] = B[boff[i] + kb + ldb];
            }
        }
    };
    auto commit = [&](int buf) {           // registers -> LDS (consumes loads)
#pragma unroll
        for (int i = 0; i < 8; ++i) Ap[buf][tid + i * 256] = ra[i];
#pragma unroll
        for (int i = 0; i < 8; ++i) {
            unsigned int v;
            int dst;
            if (TRANSB) {
                v   = rb0[i];
                dst = tid + i * 256;
            } else {
                int p = tid + i * 256;
                v   = rb0[i] | (rb1[i] << 16);
                dst = (p & 127) * 16 + (p >> 7);
            }
            if (NGUARD) v &= bmask[i];
            Bp[buf][dst] = v;
        }
    };

    const int nk = K >> 5;                 // K % 32 == 0
    fetch(0);
    commit(0);

    for (int kt = 0; kt < nk; ++kt) {
        __syncthreads();                   // tile kt visible; reads of kt-1 done
        const bool more = (kt + 1 < nk);
        if (more) fetch((kt + 1) << 5);    // loads in flight during WMMA below

        const int buf = kt & 1;
        FragAB af[2];
#pragma unroll
        for (int mi = 0; mi < 2; ++mi) {
            int r = wm * 32 + mi * 16 + l16;
            const unsigned int* p0 = &Ap[buf][r * 16 + half * 4];      // K = half*8+0..7
            const unsigned int* p1 = &Ap[buf][r * 16 + 8 + half * 4];  // K = 16+half*8+0..7
#pragma unroll
            for (int j = 0; j < 4; ++j) { af[mi].u[j] = p0[j]; af[mi].u[4 + j] = p1[j]; }
        }
        FragAB bfm[4];
#pragma unroll
        for (int ni = 0; ni < 4; ++ni) {
            int n = wn * 64 + ni * 16 + l16;
            const unsigned int* p = &Bp[buf][n * 16 + half * 4];       // K = half*8+0..7, 16+half*8+0..7
#pragma unroll
            for (int j = 0; j < 4; ++j) { bfm[ni].u[j] = p[j]; bfm[ni].u[4 + j] = p[8 + j]; }
        }
#pragma unroll
        for (int mi = 0; mi < 2; ++mi)
#pragma unroll
            for (int ni = 0; ni < 4; ++ni)
                acc[mi][ni].v = __builtin_amdgcn_wmma_f32_16x16x32_bf16(
                    false, af[mi].v, false, bfm[ni].v,
                    (short)0, acc[mi][ni].v, false, false);

        if (more) commit((kt + 1) & 1);    // wait_loadcnt lands here, after WMMA
    }

    // ---- epilogue: C/D 16x16 f32 layout (VGPR r -> M = r + 8*half) ----
#pragma unroll
    for (int mi = 0; mi < 2; ++mi)
#pragma unroll
        for (int ni = 0; ni < 4; ++ni)
#pragma unroll
            for (int r = 0; r < 8; ++r) {
                int row = row0 + wm * 32 + mi * 16 + half * 8 + r;
                int col = col0 + wn * 64 + ni * 16 + l16;
                if (!NGUARD || col < N) {
                    long idx = (long)row * ldc + col;
                    float v = alpha * acc[mi][ni].f[r];
                    if (Cin) v += Cin[idx];
                    if (act) v = gelu_t(v);
                    if (Cout) Cout[idx] = v;
                    if (Cbf)  Cbf[idx] = f2bf(v);
                }
            }
}

// ---------------------------------------------------------------------------
// Elementwise / reduction kernels
// ---------------------------------------------------------------------------
__global__ void k_f2bf(const float* __restrict__ s, unsigned short* __restrict__ d, int n) {
    int i = blockIdx.x * 256 + threadIdx.x;
    if (i < n) d[i] = f2bf(s[i]);
}

__global__ void k_zerof(float* __restrict__ p, int n) {
    int i = blockIdx.x * 256 + threadIdx.x;
    if (i < n) p[i] = 0.f;
}

__global__ void k_embed(const int* __restrict__ xtok, const float* __restrict__ emb,
                        float* __restrict__ h, unsigned short* __restrict__ hbf) {
    int s = blockIdx.x;
    int id = xtok[2 * s + 1];
    for (int d = threadIdx.x; d < D_; d += 256) {
        float v = emb[(long)id * D_ + d];
        h[s * D_ + d] = v;
        hbf[s * D_ + d] = f2bf(v);
    }
}

__global__ __launch_bounds__(256)
void k_softmax(const float* __restrict__ scores, const int* __restrict__ xtok,
               unsigned short* __restrict__ attbf) {
    int row = blockIdx.x, tid = threadIdx.x;
    __shared__ float red[256];
    const float* sr = scores + (long)row * S_;
    float vals[8];
    float mx = -3.4e38f;
#pragma unroll
    for (int i = 0; i < 8; ++i) {
        int j = tid + i * 256;
        float v = sr[j];
        v = (xtok[2 * j + 1] != 0) ? v : -1e9f;   // PAD mask
        vals[i] = v;
        mx = fmaxf(mx, v);
    }
    red[tid] = mx; __syncthreads();
    for (int st = 128; st > 0; st >>= 1) {
        if (tid < st) red[tid] = fmaxf(red[tid], red[tid + st]);
        __syncthreads();
    }
    mx = red[0]; __syncthreads();
    float sum = 0.f;
#pragma unroll
    for (int i = 0; i < 8; ++i) { vals[i] = __expf(vals[i] - mx); sum += vals[i]; }
    red[tid] = sum; __syncthreads();
    for (int st = 128; st > 0; st >>= 1) {
        if (tid < st) red[tid] += red[tid + st];
        __syncthreads();
    }
    float inv = 1.f / red[0];
#pragma unroll
    for (int i = 0; i < 8; ++i)
        attbf[(long)row * S_ + tid + i * 256] = f2bf(vals[i] * inv);
}

__global__ __launch_bounds__(256)
void k_layernorm(const float* __restrict__ x, const float* __restrict__ g,
                 const float* __restrict__ b, float* __restrict__ o) {
    int row = blockIdx.x, tid = threadIdx.x;
    __shared__ float r1[256], r2[256];
    const float* xr = x + (long)row * D_;
    float a0 = xr[tid], a1 = xr[tid + 256], a2 = xr[tid + 512];
    r1[tid] = a0 + a1 + a2;
    r2[tid] = a0 * a0 + a1 * a1 + a2 * a2;
    __syncthreads();
    for (int st = 128; st > 0; st >>= 1) {
        if (tid < st) { r1[tid] += r1[tid + st]; r2[tid] += r2[tid + st]; }
        __syncthreads();
    }
    float mean = r1[0] / (float)D_;
    float var  = r2[0] / (float)D_ - mean * mean;
    float inv  = rsqrtf(var + 1e-5f);
    float* orow = o + (long)row * D_;
    orow[tid]       = g[tid]       * (a0 - mean) * inv + b[tid];
    orow[tid + 256] = g[tid + 256] * (a1 - mean) * inv + b[tid + 256];
    orow[tid + 512] = g[tid + 512] * (a2 - mean) * inv + b[tid + 512];
}

__global__ void k_segcount(const int* __restrict__ pairs, int n, float* __restrict__ cnt) {
    int i = blockIdx.x * 256 + threadIdx.x;
    if (i < n) atomicAdd(&cnt[pairs[2 * i]], 1.f);
}

__global__ void k_segsum_tok(const int* __restrict__ xtok, const float* __restrict__ t,
                             float* __restrict__ sums) {
    int i = blockIdx.x * 256 + threadIdx.x;
    if (i < S_ * D_) {
        int s = i / D_, d = i - s * D_;
        atomicAdd(&sums[xtok[2 * s] * D_ + d], t[i]);
    }
}

__global__ void k_segsum_char(const int* __restrict__ chars, const float* __restrict__ cemb,
                              float* __restrict__ sums) {
    int i = blockIdx.x * 256 + threadIdx.x;
    if (i < CC_ * DC_) {
        int c = i / DC_, d = i - c * DC_;
        int seg = chars[2 * c], cid = chars[2 * c + 1];
        atomicAdd(&sums[seg * DC_ + d], cemb[cid * DC_ + d]);
    }
}

__global__ void k_buildcat(const float* __restrict__ sums, const float* __restrict__ cnt,
                           const float* __restrict__ sums2, const float* __restrict__ cnt2,
                           const float* __restrict__ cemb, const int* __restrict__ sosp,
                           unsigned short* __restrict__ cat) {
    int i = blockIdx.x * 256 + threadIdx.x;
    if (i < T_ * (D_ + DC_)) {
        int t = i / (D_ + DC_), d = i - t * (D_ + DC_);
        float v;
        if (d < D_) {
            v = sums[(t + 1) * D_ + d] / fmaxf(cnt[t + 1], 1.f);
        } else {
            int dc = d - D_;
            v = sums2[(t + 1) * DC_ + dc] / fmaxf(cnt2[t + 1], 1.f)
              + cemb[sosp[0] * DC_ + dc];
        }
        cat[i] = f2bf(v);
    }
}

__global__ void k_buildX(const float* __restrict__ pre, const float* __restrict__ bc,
                         const float* __restrict__ pos, unsigned short* __restrict__ X) {
    int i = blockIdx.x * 256 + threadIdx.x;
    if (i < T_ * D_) {
        int t = i / D_, d = i - t * D_;
        float c = tanhf(pre[i] + bc[d]);
#pragma unroll
        for (int l = 0; l < L_; ++l)
            X[((long)(t * L_ + l)) * D_ + d] = f2bf(c + pos[l * D_ + d]);
    }
}

// ---------------------------------------------------------------------------
// Host orchestration
// ---------------------------------------------------------------------------
extern "C" void kernel_launch(void* const* d_in, const int* in_sizes, int n_in,
                              void* d_out, int out_size, void* d_ws, size_t ws_size,
                              hipStream_t stream) {
    (void)in_sizes; (void)n_in; (void)out_size; (void)ws_size;

    const int*   xtok  = (const int*)d_in[0];
    const int*   chars = (const int*)d_in[1];
    const float* emb   = (const float*)d_in[2];
    const float* Wq    = (const float*)d_in[3];
    const float* Wk    = (const float*)d_in[4];
    const float* Wv    = (const float*)d_in[5];
    const float* Wo    = (const float*)d_in[6];
    const float* W1    = (const float*)d_in[7];
    const float* W2    = (const float*)d_in[8];
    const float* g1    = (const float*)d_in[9];
    const float* b1    = (const float*)d_in[10];
    const float* cemb  = (const float*)d_in[11];
    const float* Wc    = (const float*)d_in[12];
    const float* bc    = (const float*)d_in[13];
    const float* pos   = (const float*)d_in[14];
    const float* Wout  = (const float*)d_in[15];
    const int*   sosp  = (const int*)d_in[16];
    float*       out   = (float*)d_out;

    // Bump allocator over d_ws (256B aligned)
    char* ws = (char*)d_ws;
    size_t off = 0;
    auto alloc = [&](size_t bytes) -> void* {
        void* p = ws + off;
        off = (off + bytes + 255) & ~(size_t)255;
        return p;
    };

    typedef unsigned short u16;
    u16* wqb   = (u16*)alloc((size_t)D_ * D_ * 2);
    u16* wkb   = (u16*)alloc((size_t)D_ * D_ * 2);
    u16* wvb   = (u16*)alloc((size_t)D_ * D_ * 2);
    u16* wob   = (u16*)alloc((size_t)D_ * D_ * 2);
    u16* w1b   = (u16*)alloc((size_t)D_ * DF_ * 2);
    u16* w2b   = (u16*)alloc((size_t)DF_ * D_ * 2);
    u16* wcb   = (u16*)alloc((size_t)(D_ + DC_) * D_ * 2);
    u16* woutb = (u16*)alloc((size_t)D_ * CV_ * 2);

    float* h     = (float*)alloc((size_t)S_ * D_ * 4);
    u16*   hbf   = (u16*)  alloc((size_t)S_ * D_ * 2);
    u16*   qb    = (u16*)  alloc((size_t)S_ * D_ * 2);
    u16*   kb    = (u16*)  alloc((size_t)S_ * D_ * 2);
    u16*   vb    = (u16*)  alloc((size_t)S_ * D_ * 2);
    float* scores= (float*)alloc((size_t)S_ * S_ * 4);
    u16*   attb  = (u16*)  alloc((size_t)S_ * S_ * 2);
    u16*   attnb = (u16*)  alloc((size_t)S_ * D_ * 2);
    u16*   midgb = (u16*)  alloc((size_t)S_ * DF_ * 2);
    float* tctx  = (float*)alloc((size_t)S_ * D_ * 4);

    const int nseg = (T_ + 1) * D_ + (T_ + 1) + (T_ + 1) * DC_ + (T_ + 1);
    float* segbuf = (float*)alloc((size_t)nseg * 4);
    float* sums  = segbuf;
    float* cnt   = sums + (T_ + 1) * D_;
    float* sums2 = cnt + (T_ + 1);
    float* cnt2  = sums2 + (T_ + 1) * DC_;

    u16*   catb    = (u16*)  alloc((size_t)T_ * (D_ + DC_) * 2);
    float* combpre = (float*)alloc((size_t)T_ * D_ * 4);
    u16*   Xb      = (u16*)  alloc((size_t)T_ * L_ * D_ * 2);

    auto cvt = [&](const float* s, u16* d, int n) {
        k_f2bf<<<(n + 255) / 256, 256, 0, stream>>>(s, d, n);
    };
    auto gemm = [&](const u16* A, int lda, const u16* B, int ldb, int tB,
                    const float* Cin, float* Cout, u16* Cb, int ldc,
                    int M, int N, int K, float alpha, int act) {
        dim3 g((N + 127) / 128, (M + 127) / 128);
        if (tB)
            k_gemm_bf16<true, false><<<g, 256, 0, stream>>>(
                A, lda, B, ldb, Cin, Cout, Cb, ldc, M, N, K, alpha, act);
        else if (N % 128)
            k_gemm_bf16<false, true><<<g, 256, 0, stream>>>(
                A, lda, B, ldb, Cin, Cout, Cb, ldc, M, N, K, alpha, act);
        else
            k_gemm_bf16<false, false><<<g, 256, 0, stream>>>(
                A, lda, B, ldb, Cin, Cout, Cb, ldc, M, N, K, alpha, act);
    };

    // 1. weights -> bf16
    cvt(Wq, wqb, D_ * D_);   cvt(Wk, wkb, D_ * D_);
    cvt(Wv, wvb, D_ * D_);   cvt(Wo, wob, D_ * D_);
    cvt(W1, w1b, D_ * DF_);  cvt(W2, w2b, DF_ * D_);
    cvt(Wc, wcb, (D_ + DC_) * D_);  cvt(Wout, woutb, D_ * CV_);

    // 2. embedding gather
    k_embed<<<S_, 256, 0, stream>>>(xtok, emb, h, hbf);

    // 3. QKV projections (bf16 outputs only)
    gemm(hbf, D_, wqb, D_, 0, nullptr, nullptr, qb, D_, S_, D_, D_, 1.f, 0);
    gemm(hbf, D_, wkb, D_, 0, nullptr, nullptr, kb, D_, S_, D_, D_, 1.f, 0);
    gemm(hbf, D_, wvb, D_, 0, nullptr, nullptr, vb, D_, S_, D_, D_, 1.f, 0);

    // 4. attention, head by head (scores GEMM -> softmax -> att@V GEMM)
    for (int hh = 0; hh < H_; ++hh) {
        gemm(qb + hh * DH_, D_, kb + hh * DH_, D_, /*transB=*/1,
             nullptr, scores, nullptr, S_, S_, S_, DH_, 0.125f /* 1/sqrt(64) */, 0);
        k_softmax<<<S_, 256, 0, stream>>>(scores, xtok, attb);
        gemm(attb, S_, vb + hh * DH_, D_, 0,
             nullptr, nullptr, attnb + hh * DH_, D_, S_, DH_, S_, 1.f, 0);
    }

    // 5. output projection + residual:  h = h + attn @ Wo
    gemm(attnb, D_, wob, D_, 0, h, h, hbf, D_, S_, D_, D_, 1.f, 0);

    // 6. FFN:  h = h + gelu(h @ W1) @ W2   (gelu fused into GEMM1 epilogue)
    gemm(hbf, D_, w1b, DF_, 0, nullptr, nullptr, midgb, DF_, S_, DF_, D_, 1.f, 1);
    gemm(midgb, DF_, w2b, D_, 0, h, h, nullptr, D_, S_, D_, DF_, 1.f, 0);

    // 7. layernorm -> token context
    k_layernorm<<<S_, 256, 0, stream>>>(h, g1, b1, tctx);

    // 8. segment means (tokens and chars)
    k_zerof<<<(nseg + 255) / 256, 256, 0, stream>>>(segbuf, nseg);
    k_segcount<<<(S_ + 255) / 256, 256, 0, stream>>>(xtok, S_, cnt);
    k_segcount<<<(CC_ + 255) / 256, 256, 0, stream>>>(chars, CC_, cnt2);
    k_segsum_tok<<<(S_ * D_ + 255) / 256, 256, 0, stream>>>(xtok, tctx, sums);
    k_segsum_char<<<(CC_ * DC_ + 255) / 256, 256, 0, stream>>>(chars, cemb, sums2);

    // 9. combined = tanh([token_state, char_state] @ Wc + bc)
    k_buildcat<<<(T_ * (D_ + DC_) + 255) / 256, 256, 0, stream>>>(
        sums, cnt, sums2, cnt2, cemb, sosp, catb);
    gemm(catb, D_ + DC_, wcb, D_, 0, nullptr, combpre, nullptr, D_,
         T_, D_, D_ + DC_, 1.f, 0);

    // 10. logits = (combined + pos_emb) @ Wout  ->  d_out (f32)
    k_buildX<<<(T_ * D_ + 255) / 256, 256, 0, stream>>>(combpre, bc, pos, Xb);
    gemm(Xb, D_, woutb, CV_, 0, nullptr, out, nullptr, CV_,
         T_ * L_, CV_, D_, 1.f, 0);
}
